// FlashExpressionAttentionLayer_76922864272174
// MI455X (gfx1250) — matrix operand order repaired and checked by reading, not compiled, not run on hardware
//
#include <hip/hip_runtime.h>
#include <hip/hip_bf16.h>

// Problem constants (fixed by reference)
#define BB   4
#define SS   1024
#define DD   512
#define HH   8
#define HD   64
#define SCALE 0.125f          // HD^-0.5
#define EPSV  1e-8f
#define NEGV  -1000000000.0f

typedef __attribute__((ext_vector_type(16))) __bf16 v16bf;
typedef __attribute__((ext_vector_type(8)))  float  v8f;
typedef int v4i_ __attribute__((vector_size(16)));

union Frag16 { v16bf v; float4 f[2]; };

// ---- WMMA helpers -----------------------------------------------------------

__device__ __forceinline__ v8f wmma_bf16(v16bf a, v16bf b, v8f c) {
  return __builtin_amdgcn_wmma_f32_16x16x32_bf16(
      /*neg_a=*/false, a, /*neg_b=*/false, b,
      /*c_mod=*/(short)0, c, /*reuse_a=*/false, /*reuse_b=*/false);
}

// A-matrix (16x32 bf16, row-major, leading dim ld in elements).
// Lane L holds row M=L%16; kbase0=(L/16)*8; elems 0-7 -> K=kbase0..+7,
// elems 8-15 -> K=kbase0+16..+23   (ISA 7.12.2, 16-bit A 16x32)
__device__ __forceinline__ v16bf load_a_frag(const __bf16* A, int ld, int lane) {
  const __bf16* p = A + (size_t)(lane & 15) * (size_t)ld + ((lane >> 4) << 3);
  Frag16 r;
  r.f[0] = *(const float4*)(p);
  r.f[1] = *(const float4*)(p + 16);
  return r.v;
}

// B-matrix (32x16 bf16) fed from row-major W[n, k] (ld in elements):
// lane L holds column N=L%16 (= row n of W); kbase0=(L/16)*8; elems 0-7 ->
// K=kbase0..+7, elems 8-15 -> K=kbase0+16..+23 (same K order as A).
__device__ __forceinline__ v16bf load_b_frag(const __bf16* W, int ld, int lane) {
  const __bf16* p = W + (size_t)(lane & 15) * (size_t)ld + ((lane >> 4) << 3);
  Frag16 r;
  r.f[0] = *(const float4*)(p);
  r.f[1] = *(const float4*)(p + 16);
  return r.v;
}

// ---- CDNA5 async global->LDS (ASYNCcnt path) --------------------------------

__device__ __forceinline__ void async_copy16(const __bf16* g, __bf16* l) {
#if defined(__has_builtin) && __has_builtin(__builtin_amdgcn_global_load_async_to_lds_b128)
  typedef __attribute__((address_space(1))) v4i_ gvec;   // global int4
  typedef __attribute__((address_space(3))) v4i_ lvec;   // LDS int4
  __builtin_amdgcn_global_load_async_to_lds_b128((gvec*)g, (lvec*)l, 0, 0);
#else
  unsigned loff = (unsigned)(unsigned long long)(__attribute__((address_space(3))) char*)l;
  unsigned long long gaddr = (unsigned long long)g;
  asm volatile("global_load_async_to_lds_b128 %0, %1, off"
               :: "v"(loff), "v"(gaddr) : "memory");
#endif
}

__device__ __forceinline__ void wait_async0() {
#if defined(__has_builtin) && __has_builtin(__builtin_amdgcn_s_wait_asynccnt)
  __builtin_amdgcn_s_wait_asynccnt(0);
#else
  asm volatile("s_wait_asynccnt 0x0" ::: "memory");
#endif
}

// ---- Conversion kernels -----------------------------------------------------

__global__ void cvt_f32_bf16(const float* __restrict__ src, __bf16* __restrict__ dst, int n) {
  int i = blockIdx.x * blockDim.x + threadIdx.x;
  if (i < n) dst[i] = (__bf16)src[i];
}

// Xcat[r, c] = c < D ? gene[r, c] : expr[r, c - D]   (rows = B*S, width 2D)
__global__ void concat_bf16(const float* __restrict__ a, const float* __restrict__ b,
                            __bf16* __restrict__ dst, int total) {
  int i = blockIdx.x * blockDim.x + threadIdx.x;
  if (i >= total) return;
  int r = i / (2 * DD), c = i % (2 * DD);
  float v = (c < DD) ? a[(size_t)r * DD + c] : b[(size_t)r * DD + (c - DD)];
  dst[i] = (__bf16)v;
}

// ---- Generic WMMA GEMM:  C[m,n] = sum_k A[m,k] * W[n,k] + bias[n] -----------
// Block = 8 waves; the block's 16-row A tile is staged into LDS once with
// GLOBAL_LOAD_ASYNC_TO_LDS_B128 and shared by all 8 waves (8x A-traffic cut);
// weight B-tiles are small and L2-resident.
// MODE 0: bf16 out, row-major (ldc = N)
// MODE 1: bf16 out, Q/K head layout  (B,H,S,HD)  from (m=(b,s), n=(h,hd))
// MODE 2: bf16 out, V transposed     (B,H,HD,S)
// MODE 3: f32  out, row-major

template <int MODE>
__global__ void gemm_wmma(const __bf16* __restrict__ A, int lda,
                          const __bf16* __restrict__ W, int ldb,
                          const float* __restrict__ bias,
                          void* __restrict__ Cout,
                          int Mdim, int Ndim, int Kdim) {
  extern __shared__ char gsm[];
  __bf16* Alds = (__bf16*)gsm;               // 16 x Kdim bf16
  int lane = threadIdx.x & 31;
  int wave = threadIdx.x >> 5;
  int ngroups = Ndim >> 7;                   // groups of 8 n-tiles
  int bm = blockIdx.x / ngroups;
  int tn = (blockIdx.x % ngroups) * 8 + wave;

  // stage A tile (16 x Kdim) into LDS via async copies
  const __bf16* Abase = A + (size_t)bm * 16 * (size_t)lda;
  int cpr = Kdim >> 3;                       // 16-byte chunks per row
  for (int c = threadIdx.x; c < 16 * cpr; c += blockDim.x) {
    int row = c / cpr, col = (c % cpr) << 3;
    async_copy16(Abase + (size_t)row * lda + col, Alds + row * Kdim + col);
  }
  wait_async0();
  __syncthreads();

  const __bf16* Wb = W + (size_t)tn * 16 * (size_t)ldb;
  v8f acc = {};
  for (int kk = 0; kk < Kdim; kk += 32) {
    v16bf af = load_a_frag(Alds + kk, Kdim, lane);   // ds_load_b128 x2
    v16bf bf_ = load_b_frag(Wb + kk, ldb, lane);     // global_load_b128 x2
    acc = wmma_bf16(af, bf_, acc);
  }

  int n  = lane & 15, hi = lane >> 4;
  int gn = tn * 16 + n;
  float bv = bias[gn];
#pragma unroll
  for (int r = 0; r < 8; ++r) {
    int gm = bm * 16 + r + hi * 8;
    float v = acc[r] + bv;
    if (MODE == 0) {
      ((__bf16*)Cout)[(size_t)gm * Ndim + gn] = (__bf16)v;
    } else if (MODE == 1) {
      int b_ = gm >> 10, s_ = gm & (SS - 1), h_ = gn >> 6, d_ = gn & (HD - 1);
      ((__bf16*)Cout)[((((size_t)b_ * HH + h_) * SS) + s_) * HD + d_] = (__bf16)v;
    } else if (MODE == 2) {
      int b_ = gm >> 10, s_ = gm & (SS - 1), h_ = gn >> 6, d_ = gn & (HD - 1);
      ((__bf16*)Cout)[(((size_t)b_ * HH + h_) * HD + d_) * SS + s_] = (__bf16)v;
    } else {
      ((float*)Cout)[(size_t)gm * Ndim + gn] = v;
    }
  }
}

// ---- Attention: per-wave 16-query tile, two-pass masked softmax in LDS ------
// Q: (B,H,S,HD) bf16   K: (B,H,S,HD) bf16   Vt: (B,H,HD,S) bf16
// O: (B,S,D) bf16  (attn output before Wo projection)
#define ATTN_WAVES 2

__global__ void attn_wmma(const __bf16* __restrict__ Qh, const __bf16* __restrict__ Kh,
                          const __bf16* __restrict__ Vt, const float* __restrict__ Mmask,
                          __bf16* __restrict__ O) {
  extern __shared__ char smem[];
  int lane = threadIdx.x & 31;
  int wave = threadIdx.x >> 5;
  int t  = blockIdx.x * ATTN_WAVES + wave;   // 0 .. B*H*(S/16)-1 = 2047
  int bh = t >> 6;                           // 64 q-tiles per (b,h)
  int qt = t & 63;
  int b  = bh >> 3;
  int q0 = qt << 4;

  float* Srow = (float*)(smem + (size_t)wave * 16 * SS * sizeof(float));
  const __bf16* Qbase = Qh + ((size_t)bh * SS + q0) * HD;
  const __bf16* Kbase = Kh + (size_t)bh * SS * HD;
  const __bf16* Vbase = Vt + (size_t)bh * HD * SS;
  const float*  Mbase = Mmask + ((size_t)b * SS + q0) * SS;

  // ---- pass 1: masked scores -> LDS (fp32, 16 x 1024) ----
  for (int k0 = 0; k0 < SS; k0 += 16) {
    // hide the strided mask loads for the next tile behind this tile's WMMAs
    if (k0 + 16 < SS)
      __builtin_prefetch(Mbase + (size_t)(lane & 15) * SS + k0 + 16, 0, 1);
    v8f acc = {};
#pragma unroll
    for (int kk = 0; kk < HD; kk += 32) {
      v16bf af = load_a_frag(Qbase + kk, HD, lane);
      v16bf bf_ = load_b_frag(Kbase + (size_t)k0 * HD + kk, HD, lane);
      acc = wmma_bf16(af, bf_, acc);
    }
    int n = lane & 15, hi = lane >> 4;
#pragma unroll
    for (int r = 0; r < 8; ++r) {
      int m = r + hi * 8;
      float mv = Mbase[(size_t)m * SS + k0 + n];
      float sv = acc[r] * SCALE + ((mv == 0.0f) ? NEGV : 0.0f);
      Srow[m * SS + k0 + n] = sv;
    }
  }
  asm volatile("s_wait_dscnt 0x0" ::: "memory");

  // ---- pass 2: per-row softmax + sparse L1 renorm; bf16 P written in place --
  // masked entries forced to exactly 0 => A_bar = p / (Z*(1+eps))
  if (lane < 16) {
    float* row = Srow + lane * SS;
    float mx = -3.0e38f;
    for (int j = 0; j < SS; ++j) mx = fmaxf(mx, row[j]);
    float z = 0.0f;
    for (int j = 0; j < SS; ++j) {
      float s = row[j];
      float p = (s > 0.5f * NEGV) ? __expf(s - mx) : 0.0f;
      row[j] = p;
      z += p;
    }
    float inv = 1.0f / (z * (1.0f + EPSV) + 1e-30f);
    __bf16* prow = (__bf16*)row;   // in-place: write@2j bytes <= read@4j bytes
    for (int j = 0; j < SS; ++j) prow[j] = (__bf16)(row[j] * inv);
  }
  asm volatile("s_wait_dscnt 0x0" ::: "memory");

  // ---- pass 3: O(16 x 64) = P(16 x 1024, LDS bf16, row stride 2048) @ Vt^T --
  const __bf16* P = (const __bf16*)Srow;   // bf16 view, row stride 2048 elems
  v8f oacc[4] = {{}, {}, {}, {}};
  for (int kk = 0; kk < SS; kk += 32) {
    v16bf af = load_a_frag(P + kk, 2048, lane);   // ds_load_b128 x2
#pragma unroll
    for (int nt = 0; nt < 4; ++nt) {
      v16bf bf_ = load_b_frag(Vbase + (size_t)(nt * 16) * SS + kk, SS, lane);
      oacc[nt] = wmma_bf16(af, bf_, oacc[nt]);
    }
  }

  int n = lane & 15, hi = lane >> 4;
  int h = bh & (HH - 1);
#pragma unroll
  for (int nt = 0; nt < 4; ++nt) {
#pragma unroll
    for (int r = 0; r < 8; ++r) {
      int m = r + hi * 8;
      size_t idx = ((size_t)b * SS + (q0 + m)) * DD + h * HD + nt * 16 + n;
      O[idx] = (__bf16)oacc[nt][r];
    }
  }
}

// ---- Host-side orchestration ------------------------------------------------

extern "C" void kernel_launch(void* const* d_in, const int* in_sizes, int n_in,
                              void* d_out, int out_size, void* d_ws, size_t ws_size,
                              hipStream_t stream) {
  const float* gene = (const float*)d_in[0];   // (B,S,D)
  const float* expr = (const float*)d_in[1];   // (B,S,D)
  const float* Mm   = (const float*)d_in[2];   // (B,S,S)
  const float* Wf   = (const float*)d_in[3];   // (D,2D)
  const float* bf_  = (const float*)d_in[4];
  const float* Wq   = (const float*)d_in[5];
  const float* bq   = (const float*)d_in[6];
  const float* Wk   = (const float*)d_in[7];
  const float* bk   = (const float*)d_in[8];
  const float* Wv   = (const float*)d_in[9];
  const float* bv   = (const float*)d_in[10];
  const float* Wo   = (const float*)d_in[11];
  const float* bo   = (const float*)d_in[12];
  float* out = (float*)d_out;

  const int ROWS = BB * SS;                    // 4096
  char* ws = (char*)d_ws;
  size_t off = 0;
  auto carve = [&](size_t bytes) -> char* {
    char* p = ws + off;
    off = (off + bytes + 255) & ~(size_t)255;
    return p;
  };
  __bf16* Xcat = (__bf16*)carve((size_t)ROWS * 2 * DD * 2);  // 8 MB
  __bf16* Ebf  = (__bf16*)carve((size_t)ROWS * DD * 2);      // 4 MB
  __bf16* Wfb  = (__bf16*)carve((size_t)DD * 2 * DD * 2);    // 1 MB
  __bf16* Wqb  = (__bf16*)carve((size_t)DD * DD * 2);
  __bf16* Wkb  = (__bf16*)carve((size_t)DD * DD * 2);
  __bf16* Wvb  = (__bf16*)carve((size_t)DD * DD * 2);
  __bf16* Wob  = (__bf16*)carve((size_t)DD * DD * 2);
  __bf16* Fbf  = (__bf16*)carve((size_t)ROWS * DD * 2);      // fused, bf16
  __bf16* Qh   = (__bf16*)carve((size_t)ROWS * DD * 2);
  __bf16* Kh   = (__bf16*)carve((size_t)ROWS * DD * 2);
  __bf16* Vt   = (__bf16*)carve((size_t)ROWS * DD * 2);
  __bf16* AO   = (__bf16*)carve((size_t)ROWS * DD * 2);
  (void)ws_size; (void)in_sizes; (void)n_in; (void)out_size;

  const int T = 256;
  auto blk = [](int n, int t) { return (n + t - 1) / t; };

  // 1) conversions
  concat_bf16<<<blk(ROWS * 2 * DD, T), T, 0, stream>>>(gene, expr, Xcat, ROWS * 2 * DD);
  cvt_f32_bf16<<<blk(ROWS * DD, T), T, 0, stream>>>(expr, Ebf, ROWS * DD);
  cvt_f32_bf16<<<blk(DD * 2 * DD, T), T, 0, stream>>>(Wf, Wfb, DD * 2 * DD);
  cvt_f32_bf16<<<blk(DD * DD, T), T, 0, stream>>>(Wq, Wqb, DD * DD);
  cvt_f32_bf16<<<blk(DD * DD, T), T, 0, stream>>>(Wk, Wkb, DD * DD);
  cvt_f32_bf16<<<blk(DD * DD, T), T, 0, stream>>>(Wv, Wvb, DD * DD);
  cvt_f32_bf16<<<blk(DD * DD, T), T, 0, stream>>>(Wo, Wob, DD * DD);

  // 2) projections: block = 8 waves covering 8 n-tiles of one 16-row m-tile,
  //    A tile async-staged in LDS (16*K*2 bytes dynamic LDS)
  int gblocks = (ROWS / 16) * (DD / 128);      // 1024 blocks (exact fit)
  gemm_wmma<0><<<gblocks, T, 16 * 2 * DD * 2, stream>>>(Xcat, 2 * DD, Wfb, 2 * DD, bf_, Fbf, ROWS, DD, 2 * DD);
  gemm_wmma<1><<<gblocks, T, 16 * DD * 2, stream>>>(Fbf, DD, Wqb, DD, bq, Qh, ROWS, DD, DD);
  gemm_wmma<1><<<gblocks, T, 16 * DD * 2, stream>>>(Fbf, DD, Wkb, DD, bk, Kh, ROWS, DD, DD);
  gemm_wmma<2><<<gblocks, T, 16 * DD * 2, stream>>>(Ebf, DD, Wvb, DD, bv, Vt, ROWS, DD, DD);

  // 3) attention: 2048 q-tiles, 2 waves/block, 128 KB dynamic LDS
  int qtiles = BB * HH * (SS / 16);            // 2048
  size_t lds = (size_t)ATTN_WAVES * 16 * SS * sizeof(float);
  attn_wmma<<<qtiles / ATTN_WAVES, ATTN_WAVES * 32, lds, stream>>>(Qh, Kh, Vt, Mm, AO);

  // 4) output projection -> fp32
  gemm_wmma<3><<<gblocks, T, 16 * DD * 2, stream>>>(AO, DD, Wob, DD, bo, out, ROWS, DD, DD);
}
